// MultiHeadAttention_SHORT_26628797235509
// MI455X (gfx1250) — compile-verified, matrix operand order checked by reading, not run on hardware
//
#include <hip/hip_runtime.h>

typedef __attribute__((ext_vector_type(16))) __bf16 bf16x16;
typedef __attribute__((ext_vector_type(8)))  float  f32x8;
typedef int v4i_gv __attribute__((__vector_size__(16)));

#define S_LEN 2048
#define DM    1024
#define NH    16
#define DK    64

// ---- CDNA5 async global->LDS copy (ASYNCcnt) with safe fallback -----------
#if defined(__HIP_DEVICE_COMPILE__) && __has_builtin(__builtin_amdgcn_global_load_async_to_lds_b128)
#define HAVE_ASYNC_COPY 1
__device__ __forceinline__ void async_copy_b128(const void* g, void* l) {
  __builtin_amdgcn_global_load_async_to_lds_b128(
      (__attribute__((address_space(1))) v4i_gv*)(unsigned long long)g,
      (__attribute__((address_space(3))) v4i_gv*)(unsigned int)(unsigned long long)l,
      0, 0);
}
#if __has_builtin(__builtin_amdgcn_s_wait_asynccnt)
#define WAIT_ASYNC() __builtin_amdgcn_s_wait_asynccnt(0)
#else
#define WAIT_ASYNC() asm volatile("s_wait_asynccnt 0x0" ::: "memory")
#endif
#else
#define HAVE_ASYNC_COPY 0
#define WAIT_ASYNC()
#endif

// Build a 16-element bf16 fragment covering a 32-wide K slice starting at `row`,
// per the CDNA5 16-bit A/B VGPR layout: lane<16 holds K={0..7,16..23},
// lane>=16 holds K={8..15,24..31} (kb = 0 or 8). Lowers to 2x ds_load_b128.
__device__ __forceinline__ bf16x16 load_frag32(const __bf16* row, int kb) {
  bf16x16 f;
#pragma unroll
  for (int j = 0; j < 8; ++j) {
    f[j]     = row[kb + j];
    f[8 + j] = row[16 + kb + j];
  }
  return f;
}

// ---------------------------------------------------------------------------
// GEMM: Y = X @ W^T + bias. 128x128 tile / 256 threads (8 waves),
// double-buffered LDS (one barrier per K-step), batched fragment loads.
// mode 0: X f32, write bf16 per-head [B,H,S,64]   (Q and K projections)
// mode 2: X f32, write bf16 per-head [B,H,64,S]   (V projection, transposed)
// mode 3: X bf16 (Ctx), write f32 [4096,1024]     (output projection)
// ---------------------------------------------------------------------------
__global__ void __launch_bounds__(256) mha_proj_kernel(
    const void* __restrict__ Xv, const float* __restrict__ W,
    const float* __restrict__ bias, void* __restrict__ Yout, int mode)
{
  __shared__ __bf16 As[2][128][40];   // stride 40 -> 80B rows (16B aligned)
  __shared__ __bf16 Bs[2][128][40];   // [n][k] = W[n0+n][k0+k]

  const int tid  = threadIdx.x;
  const int lane = tid & 31;
  const int wv   = tid >> 5;
  const int ln16 = lane & 15;
  const int kb   = (lane < 16) ? 0 : 8;
  const int roff = (lane < 16) ? 0 : 8;

  const int m0 = blockIdx.x * 128;
  const int n0 = blockIdx.y * 128;

  auto stage = [&](int buf, int k0) {
    if (mode != 3) {
      const float* X = (const float*)Xv;
      for (int t = tid; t < 1024; t += 256) {           // 128 rows x 8 float4
        int r = t >> 3, c = (t & 7) * 4;
        float4 v = *(const float4*)(X + (size_t)(m0 + r) * DM + k0 + c);
        As[buf][r][c]     = (__bf16)v.x; As[buf][r][c + 1] = (__bf16)v.y;
        As[buf][r][c + 2] = (__bf16)v.z; As[buf][r][c + 3] = (__bf16)v.w;
      }
    } else {
      const __bf16* X = (const __bf16*)Xv;
      for (int t = tid; t < 512; t += 256) {            // 128 rows x 4 uint4
        int r = t >> 2, c = (t & 3) * 8;
        *(uint4*)&As[buf][r][c] = *(const uint4*)(X + (size_t)(m0 + r) * DM + k0 + c);
      }
    }
    for (int t = tid; t < 1024; t += 256) {             // W tile, rows = out cols
      int r = t >> 3, c = (t & 7) * 4;
      float4 v = *(const float4*)(W + (size_t)(n0 + r) * DM + k0 + c);
      Bs[buf][r][c]     = (__bf16)v.x; Bs[buf][r][c + 1] = (__bf16)v.y;
      Bs[buf][r][c + 2] = (__bf16)v.z; Bs[buf][r][c + 3] = (__bf16)v.w;
    }
  };

  const f32x8 zero8 = {};
  f32x8 acc[8];
#pragma unroll
  for (int nt = 0; nt < 8; ++nt) acc[nt] = zero8;

  stage(0, 0);
  __syncthreads();
  for (int kt = 0; kt < DM / 32; ++kt) {
    int cur = kt & 1;
    if (kt + 1 < DM / 32) stage(cur ^ 1, (kt + 1) * 32);  // overlap next tile

    const __bf16* arow = &As[cur][wv * 16 + ln16][0];
    bf16x16 af = load_frag32(arow, kb);
    bf16x16 bfr[8];
#pragma unroll
    for (int nt = 0; nt < 8; ++nt)
      bfr[nt] = load_frag32(&Bs[cur][nt * 16 + ln16][0], kb);
#pragma unroll
    for (int nt = 0; nt < 8; ++nt)
      acc[nt] = __builtin_amdgcn_wmma_f32_16x16x32_bf16(
          false, af, false, bfr[nt], (short)0, acc[nt], false, false);
    __syncthreads();
  }

  // epilogue: bias + scatter per mode
#pragma unroll
  for (int nt = 0; nt < 8; ++nt) {
    int n = n0 + nt * 16 + ln16;
    float bval = bias[n];
#pragma unroll
    for (int i = 0; i < 8; ++i) {
      int r = m0 + wv * 16 + roff + i;       // global row in [0,4096)
      float val = acc[nt][i] + bval;
      if (mode == 3) {
        ((float*)Yout)[(size_t)r * DM + n] = val;
      } else {
        int bb = r >> 11, s = r & (S_LEN - 1);
        int hh = n >> 6,  d = n & (DK - 1);
        if (mode == 2)   // V: [B,H,64,S] (d-major for P@V B-fragments)
          ((__bf16*)Yout)[(((size_t)(bb * NH + hh)) * DK + d) * S_LEN + s] = (__bf16)val;
        else             // Q/K: [B,H,S,64]
          ((__bf16*)Yout)[(((size_t)(bb * NH + hh)) * S_LEN + s) * DK + d] = (__bf16)val;
      }
    }
  }
}

// ---------------------------------------------------------------------------
// Flash attention: grid (S/128 q-tiles, B*H). 256 threads = 8 waves,
// 16 queries per wave, double-buffered async-staged 64-key blocks,
// online softmax with 16-lane butterfly reductions.
// ---------------------------------------------------------------------------
__global__ void __launch_bounds__(256) mha_flash_kernel(
    const __bf16* __restrict__ Qp, const __bf16* __restrict__ Kp,
    const __bf16* __restrict__ Vp, const int* __restrict__ mask,
    __bf16* __restrict__ Ctx)
{
  __shared__ __bf16 Ks[2][64][72];     // [key][d]
  __shared__ __bf16 Vt[2][64][72];     // [d][key]
  __shared__ __bf16 Pb[8][16][72];     // per-wave P tile (C->A relayout)

  const int tid  = threadIdx.x;
  const int lane = tid & 31;
  const int wv   = tid >> 5;
  const int ln16 = lane & 15;
  const int kb   = (lane < 16) ? 0 : 8;
  const int roff = (lane < 16) ? 0 : 8;

  const int q0 = blockIdx.x * 128;
  const int bh = blockIdx.y;
  const int b  = bh >> 4;
  const int h  = bh & 15;

  const __bf16* Kbase = Kp + (size_t)bh * S_LEN * DK;
  const __bf16* Vbase = Vp + (size_t)bh * DK * S_LEN;
  const int*    mbase = mask + (size_t)b * S_LEN * S_LEN;

  auto stage = [&](int buf, int blk) {   // 64x64 bf16 K tile + V^T tile
    for (int t = tid; t < 512; t += 256) {
      int r = t >> 3, c = (t & 7) * 8;
#if HAVE_ASYNC_COPY
      async_copy_b128(Kbase + (size_t)(blk + r) * DK + c, &Ks[buf][r][c]);
      async_copy_b128(Vbase + (size_t)r * S_LEN + blk + c, &Vt[buf][r][c]);
#else
      *(uint4*)&Ks[buf][r][c] = *(const uint4*)(Kbase + (size_t)(blk + r) * DK + c);
      *(uint4*)&Vt[buf][r][c] = *(const uint4*)(Vbase + (size_t)r * S_LEN + blk + c);
#endif
    }
  };

  // persistent Q A-fragments (16 queries x d=64 -> two K=32 fragments)
  const __bf16* qrow = Qp + (size_t)(bh * S_LEN + q0 + wv * 16 + ln16) * DK;
  bf16x16 aq0 = load_frag32(qrow, kb);
  bf16x16 aq1 = load_frag32(qrow + 32, kb);

  const f32x8 zero8 = {};
  f32x8 acc[4];
#pragma unroll
  for (int dt = 0; dt < 4; ++dt) acc[dt] = zero8;
  float mrow[8], lrow[8];
#pragma unroll
  for (int i = 0; i < 8; ++i) { mrow[i] = -3.0e38f; lrow[i] = 0.f; }

  stage(0, 0);
  for (int t = 0; t < S_LEN / 64; ++t) {
    int cur = t & 1;
    int kb0 = t * 64;
    WAIT_ASYNC();                        // this wave's staged tile is in LDS
    __syncthreads();                     // tile `cur` visible; prev reads done
    if (t + 1 < S_LEN / 64) stage(cur ^ 1, kb0 + 64);

    // S = Q K^T : 16x64 per wave. Batch all B-fragments, then WMMA chain.
    bf16x16 bk[8];
#pragma unroll
    for (int nt = 0; nt < 4; ++nt) {
      const __bf16* kr = &Ks[cur][nt * 16 + ln16][0];
      bk[2 * nt]     = load_frag32(kr, kb);
      bk[2 * nt + 1] = load_frag32(kr + 32, kb);
    }
    f32x8 st[4];
#pragma unroll
    for (int nt = 0; nt < 4; ++nt) {
      f32x8 s = zero8;
      s = __builtin_amdgcn_wmma_f32_16x16x32_bf16(false, aq0, false, bk[2 * nt],     (short)0, s, false, false);
      s = __builtin_amdgcn_wmma_f32_16x16x32_bf16(false, aq1, false, bk[2 * nt + 1], (short)0, s, false, false);
      st[nt] = s;
    }
    // scale 1/sqrt(64) then mask -> -1e9
#pragma unroll
    for (int nt = 0; nt < 4; ++nt) {
      int kcol = kb0 + nt * 16 + ln16;
#pragma unroll
      for (int i = 0; i < 8; ++i) {
        int qr = q0 + wv * 16 + roff + i;
        int mv = mbase[(size_t)qr * S_LEN + kcol];
        float sv = st[nt][i] * 0.125f;
        st[nt][i] = (mv == 0) ? -1.0e9f : sv;
      }
    }
    // online softmax: row stats via 16-lane butterflies (rows live in lane halves)
#pragma unroll
    for (int i = 0; i < 8; ++i) {
      float rm = fmaxf(fmaxf(st[0][i], st[1][i]), fmaxf(st[2][i], st[3][i]));
#pragma unroll
      for (int off = 8; off >= 1; off >>= 1) rm = fmaxf(rm, __shfl_xor(rm, off, 32));
      float mnew = fmaxf(mrow[i], rm);
      float corr = __expf(mrow[i] - mnew);
      float rs = 0.f;
#pragma unroll
      for (int nt = 0; nt < 4; ++nt) {
        float p = __expf(st[nt][i] - mnew);
        st[nt][i] = p;
        rs += p;
      }
#pragma unroll
      for (int off = 8; off >= 1; off >>= 1) rs += __shfl_xor(rs, off, 32);
      lrow[i] = lrow[i] * corr + rs;
      mrow[i] = mnew;
#pragma unroll
      for (int dt = 0; dt < 4; ++dt) acc[dt][i] *= corr;
    }
    // relayout P (C layout) -> LDS -> A fragments
#pragma unroll
    for (int nt = 0; nt < 4; ++nt)
#pragma unroll
      for (int i = 0; i < 8; ++i)
        Pb[wv][roff + i][nt * 16 + ln16] = (__bf16)st[nt][i];
    __syncthreads();

    // ctx += P @ V : batch B-fragments, then WMMA chain
    const __bf16* prow = &Pb[wv][ln16][0];
    bf16x16 ap0 = load_frag32(prow, kb);
    bf16x16 ap1 = load_frag32(prow + 32, kb);
    bf16x16 bv[8];
#pragma unroll
    for (int dt = 0; dt < 4; ++dt) {
      const __bf16* vr = &Vt[cur][dt * 16 + ln16][0];
      bv[2 * dt]     = load_frag32(vr, kb);
      bv[2 * dt + 1] = load_frag32(vr + 32, kb);
    }
#pragma unroll
    for (int dt = 0; dt < 4; ++dt) {
      acc[dt] = __builtin_amdgcn_wmma_f32_16x16x32_bf16(false, ap0, false, bv[2 * dt],     (short)0, acc[dt], false, false);
      acc[dt] = __builtin_amdgcn_wmma_f32_16x16x32_bf16(false, ap1, false, bv[2 * dt + 1], (short)0, acc[dt], false, false);
    }
  }

  // normalize and write ctx back in [B,S,H*64] layout (bf16) for the O-proj
#pragma unroll
  for (int dt = 0; dt < 4; ++dt) {
    int dcol = dt * 16 + ln16;
#pragma unroll
    for (int i = 0; i < 8; ++i) {
      int qr = q0 + wv * 16 + roff + i;
      float val = acc[dt][i] / lrow[i];
      Ctx[((size_t)b * S_LEN + qr) * DM + h * 64 + dcol] = (__bf16)val;
    }
  }
}

// ---------------------------------------------------------------------------
extern "C" void kernel_launch(void* const* d_in, const int* in_sizes, int n_in,
                              void* d_out, int out_size, void* d_ws, size_t ws_size,
                              hipStream_t stream) {
  const float* q    = (const float*)d_in[0];
  const float* k    = (const float*)d_in[1];
  const float* v    = (const float*)d_in[2];
  const int*   mask = (const int*)d_in[3];
  const float* Wq   = (const float*)d_in[4];
  const float* bq   = (const float*)d_in[5];
  const float* Wk   = (const float*)d_in[6];
  const float* bk   = (const float*)d_in[7];
  const float* Wv   = (const float*)d_in[8];
  const float* bv   = (const float*)d_in[9];
  const float* Wo   = (const float*)d_in[10];
  const float* bo   = (const float*)d_in[11];

  char* ws = (char*)d_ws;
  __bf16* Qp  = (__bf16*)(ws);                          // 8 MiB each
  __bf16* Kp  = (__bf16*)(ws + (size_t)8  * 1024 * 1024);
  __bf16* Vp  = (__bf16*)(ws + (size_t)16 * 1024 * 1024);
  __bf16* Ctx = (__bf16*)(ws + (size_t)24 * 1024 * 1024);

  dim3 pgrid(32, 8);   // 4096/128 x 1024/128
  mha_proj_kernel<<<pgrid, 256, 0, stream>>>(q, Wq, bq, Qp, 0);
  mha_proj_kernel<<<pgrid, 256, 0, stream>>>(k, Wk, bk, Kp, 0);
  mha_proj_kernel<<<pgrid, 256, 0, stream>>>(v, Wv, bv, Vp, 2);

  dim3 fgrid(16, 32);  // 2048/128 q-tiles x (B*H)
  mha_flash_kernel<<<fgrid, 256, 0, stream>>>(Qp, Kp, Vp, mask, Ctx);

  mha_proj_kernel<<<pgrid, 256, 0, stream>>>(Ctx, Wo, bo, d_out, 3);
}
